// Cross_Attention_84911503442240
// MI455X (gfx1250) — matrix operand order repaired and checked by reading, not compiled, not run on hardware
//
#include <hip/hip_runtime.h>

// Dual-gated cross attention for MI455X (gfx1250, wave32, WMMA + async-to-LDS).
// out[b,l,h,d] = sum_s tanh(scale*Q1K1^T) * sigmoid(scale*Q2K2^T) * V1
// B=2 L=2048 S=2048 H=16 E=64 D=64, fp32 in/out, f16 WMMA with f32 accum.
//
// Phase 1 (pre-pass, once): pack K1,K2 -> f16 [b,h,s,e]; V1 -> f16 [b,h,d,s] in d_ws.
// Phase 2 (main): flash-style loop, tiles staged via global_load_async_to_lds_b128
// with double buffering (ASYNCcnt), all matmuls on v_wmma_f32_16x16x32_f16.

typedef __attribute__((ext_vector_type(16))) _Float16 v16h;
typedef __attribute__((ext_vector_type(8)))  _Float16 v8h;
typedef __attribute__((ext_vector_type(4)))  _Float16 v4h;
typedef __attribute__((ext_vector_type(8)))  float    v8f;

#define BB 2
#define LL 2048
#define SS 2048
#define HH 16
#define EE 64
#define DD 64
#define LTILE 64   // L rows per block (16 per wave, 4 waves)
#define STILE 32   // S columns per inner iteration (= WMMA f16 K depth)

__device__ __forceinline__ float fast_sigmoid(float x) {
  return __builtin_amdgcn_rcpf(1.0f + __builtin_amdgcn_exp2f(-1.442695041f * x));
}

__device__ __forceinline__ void load8f(const float* __restrict__ p, float* v) {
  const float4* p4 = (const float4*)p;
  float4 a = p4[0], b = p4[1];
  v[0]=a.x; v[1]=a.y; v[2]=a.z; v[3]=a.w;
  v[4]=b.x; v[5]=b.y; v[6]=b.z; v[7]=b.w;
}

// Async global->LDS copy, 16 bytes per lane. LDS dest offset in a VGPR (VDST),
// 64-bit global address in a VGPR pair. Tracked by ASYNCcnt.
__device__ __forceinline__ void async_b128(unsigned ldsoff, const _Float16* g) {
  asm volatile("global_load_async_to_lds_b128 %0, %1, off"
               :: "v"(ldsoff), "v"(g) : "memory");
}
__device__ __forceinline__ void wait_async0() {
  asm volatile("s_wait_asynccnt 0x0" ::: "memory");
}
__device__ __forceinline__ unsigned lds_off(const _Float16* p) {
  return (unsigned)(uintptr_t)p;  // generic addr low 32 bits = LDS byte offset
}

// ---------------- Phase 1a: K1/K2 -> f16, [b,s,h,e] -> [b,h,s,e] ----------------
__global__ __launch_bounds__(256)
void pack_k_kernel(const float* __restrict__ k1, const float* __restrict__ k2,
                   _Float16* __restrict__ k1h, _Float16* __restrict__ k2h) {
  const int idx = blockIdx.x * 256 + threadIdx.x;   // chunk of 8 e-values
  const int e8 = idx & 7;
  const int s  = (idx >> 3) & (SS - 1);
  const int h  = (idx >> 14) & (HH - 1);
  const int b  = idx >> 18;
  const size_t gin  = (((size_t)b * SS + s) * HH + h) * EE + e8 * 8;
  const size_t gout = (((size_t)b * HH + h) * SS + s) * EE + e8 * 8;
  float t[8];
  load8f(k1 + gin, t);
  v8h o1;
  #pragma unroll
  for (int j = 0; j < 8; ++j) o1[j] = (_Float16)t[j];
  *(v8h*)(k1h + gout) = o1;
  load8f(k2 + gin, t);
  v8h o2;
  #pragma unroll
  for (int j = 0; j < 8; ++j) o2[j] = (_Float16)t[j];
  *(v8h*)(k2h + gout) = o2;
}

// ---------------- Phase 1b: V1 -> f16 transposed, [b,s,h,d] -> [b,h,d,s] ----------------
__global__ __launch_bounds__(256)
void pack_v_kernel(const float* __restrict__ v1, _Float16* __restrict__ vth) {
  __shared__ _Float16 tile[64][68];   // padded rows to spread LDS banks
  const int st = blockIdx.x & 31;           // S/64
  const int h  = (blockIdx.x >> 5) & 15;
  const int b  = blockIdx.x >> 9;
  const int s0 = st * 64;
  const int t  = threadIdx.x;
  #pragma unroll
  for (int it = 0; it < 4; ++it) {
    const int s = (t >> 4) + 16 * it;
    const int d = (t & 15) * 4;
    const float4 x = *(const float4*)(v1 + (((size_t)b * SS + s0 + s) * HH + h) * DD + d);
    v4h o; o[0] = (_Float16)x.x; o[1] = (_Float16)x.y; o[2] = (_Float16)x.z; o[3] = (_Float16)x.w;
    *(v4h*)&tile[s][d] = o;
  }
  __syncthreads();
  #pragma unroll
  for (int it = 0; it < 4; ++it) {
    const int d  = (t >> 4) + 16 * it;
    const int s4 = (t & 15) * 4;
    v4h o;
    #pragma unroll
    for (int j = 0; j < 4; ++j) o[j] = tile[s4 + j][d];
    *(v4h*)(vth + (((size_t)b * HH + h) * DD + d) * SS + s0 + s4) = o;
  }
}

// ---------------- Phase 2: attention main loop ----------------
__global__ __launch_bounds__(128)
void dual_attn_kernel(const float* __restrict__ q1, const float* __restrict__ q2,
                      const _Float16* __restrict__ k1h, const _Float16* __restrict__ k2h,
                      const _Float16* __restrict__ vth, float* __restrict__ out) {
  __shared__ __align__(16) _Float16 K1s[2][STILE * EE];   // [s][e] contiguous, 4KB each
  __shared__ __align__(16) _Float16 K2s[2][STILE * EE];
  __shared__ __align__(16) _Float16 Vts[2][DD * STILE];   // [d][s] transposed
  __shared__ __align__(16) _Float16 Pb[4 * 16 * STILE];   // per-wave C->A relayout tile

  const int tid  = threadIdx.x;
  const int lane = tid & 31;
  const int w    = tid >> 5;
  const int n    = lane & 15;
  const int hi   = lane >> 4;

  const int nLT = LL / LTILE;
  const int lt  = blockIdx.x % nLT;
  const int h   = (blockIdx.x / nLT) % HH;
  const int b   = blockIdx.x / (nLT * HH);
  const int l0  = lt * LTILE;
  const int bh  = b * HH + h;

  const _Float16* kt1 = k1h + (size_t)bh * SS * EE;
  const _Float16* kt2 = k2h + (size_t)bh * SS * EE;
  const _Float16* vt  = vth + (size_t)bh * DD * SS;

  // ---- Q1/Q2 A-fragments (built once from f32 global) ----
  v16h qa1[2], qa2[2];
  {
    const int ql    = l0 + w * 16 + n;
    const int qbase = ((b * LL + ql) * HH + h) * EE;
    #pragma unroll
    for (int c = 0; c < 2; ++c) {
      const int e0 = 32 * c + hi * 16;
      float t0[8], t1[8];
      load8f(q1 + qbase + e0, t0);
      load8f(q1 + qbase + e0 + 8, t1);
      v16h A1;
      #pragma unroll
      for (int j = 0; j < 8; ++j) { A1[j] = (_Float16)t0[j]; A1[8 + j] = (_Float16)t1[j]; }
      qa1[c] = A1;
      load8f(q2 + qbase + e0, t0);
      load8f(q2 + qbase + e0 + 8, t1);
      v16h A2;
      #pragma unroll
      for (int j = 0; j < 8; ++j) { A2[j] = (_Float16)t0[j]; A2[8 + j] = (_Float16)t1[j]; }
      qa2[c] = A2;
    }
  }

  v8f oacc[4];
  #pragma unroll
  for (int dg = 0; dg < 4; ++dg) oacc[dg] = (v8f){};

  // issue async copies for one S-tile into buffer `buf` (6 x b128 per thread)
  const int vd = tid >> 1;              // 0..63
  const int vso = (tid & 1) * 16;       // 0 / 16
  auto stage = [&](int s0, int buf) {
    const _Float16* g1 = kt1 + s0 * EE + tid * 16;   // K tile: contiguous 4KB
    const _Float16* g2 = kt2 + s0 * EE + tid * 16;
    const _Float16* gv = vt + (size_t)vd * SS + s0 + vso;
    const unsigned l1 = lds_off(&K1s[buf][tid * 16]);
    const unsigned l2 = lds_off(&K2s[buf][tid * 16]);
    const unsigned lv = lds_off(&Vts[buf][vd * STILE + vso]);
    async_b128(l1,      g1);
    async_b128(l1 + 16, g1 + 8);
    async_b128(l2,      g2);
    async_b128(l2 + 16, g2 + 8);
    async_b128(lv,      gv);
    async_b128(lv + 16, gv + 8);
  };

  stage(0, 0);
  for (int it = 0; it < SS / STILE; ++it) {
    wait_async0();      // this wave's copies for buffer it&1 have landed in LDS
    __syncthreads();    // all waves' copies visible; prev buffer free for reuse
    if (it + 1 < SS / STILE) stage((it + 1) * STILE, (it + 1) & 1);
    const int buf = it & 1;

    // ---- scores + gate, one 16x16 N-half at a time ----
    #pragma unroll
    for (int t = 0; t < 2; ++t) {
      v8f s1 = (v8f){};
      v8f s2 = (v8f){};
      #pragma unroll
      for (int c = 0; c < 2; ++c) {
        const v16h kb1 = *(const v16h*)&K1s[buf][(16 * t + n) * EE + 32 * c + hi * 16];
        const v16h kb2 = *(const v16h*)&K2s[buf][(16 * t + n) * EE + 32 * c + hi * 16];
        s1 = __builtin_amdgcn_wmma_f32_16x16x32_f16(false, qa1[c], false, kb1,
                                                    (short)0, s1, false, false);
        s2 = __builtin_amdgcn_wmma_f32_16x16x32_f16(false, qa2[c], false, kb2,
                                                    (short)0, s2, false, false);
      }
      #pragma unroll
      for (int r = 0; r < 8; ++r) {
        const float x1 = s1[r] * 0.125f;
        const float x2 = s2[r] * 0.125f;
        const float th = 2.0f * fast_sigmoid(2.0f * x1) - 1.0f;  // tanh(x1)
        const float a  = th * fast_sigmoid(x2);
        Pb[w * 512 + (hi * 8 + r) * STILE + 16 * t + n] = (_Float16)a;
      }
    }

    // ---- C->A relayout through wave-private LDS, then A(16x32) x V(32x64) ----
    v16h Af;
    {
      const v8h p0 = *(const v8h*)&Pb[w * 512 + n * STILE + hi * 16];
      const v8h p1 = *(const v8h*)&Pb[w * 512 + n * STILE + hi * 16 + 8];
      #pragma unroll
      for (int j = 0; j < 8; ++j) { Af[j] = p0[j]; Af[8 + j] = p1[j]; }
    }
    #pragma unroll
    for (int dg = 0; dg < 4; ++dg) {
      const v16h vb = *(const v16h*)&Vts[buf][(dg * 16 + n) * STILE + hi * 16];
      oacc[dg] = __builtin_amdgcn_wmma_f32_16x16x32_f16(false, Af, false, vb,
                                                        (short)0, oacc[dg], false, false);
    }
  }

  #pragma unroll
  for (int dg = 0; dg < 4; ++dg) {
    #pragma unroll
    for (int r = 0; r < 8; ++r) {
      const int l = l0 + w * 16 + hi * 8 + r;
      const int d = dg * 16 + n;
      out[((b * LL + l) * HH + h) * DD + d] = oacc[dg][r];
    }
  }
}

extern "C" void kernel_launch(void* const* d_in, const int* in_sizes, int n_in,
                              void* d_out, int out_size, void* d_ws, size_t ws_size,
                              hipStream_t stream) {
  const float* q1 = (const float*)d_in[0];
  const float* k1 = (const float*)d_in[1];
  const float* v1 = (const float*)d_in[2];
  const float* q2 = (const float*)d_in[3];
  const float* k2 = (const float*)d_in[4];
  // d_in[5] = v2 (unused by reference), d_in[6] = attn_mask (unused)
  float* out = (float*)d_out;

  const size_t nkv = (size_t)BB * HH * SS * EE;  // 4,194,304 elems
  _Float16* k1h = (_Float16*)d_ws;
  _Float16* k2h = k1h + nkv;
  _Float16* vth = k2h + nkv;                     // total 24 MB of d_ws

  pack_k_kernel<<<dim3((BB * HH * SS * (EE / 8)) / 256), dim3(256), 0, stream>>>(k1, k2, k1h, k2h);
  pack_v_kernel<<<dim3(BB * HH * (SS / 64)), dim3(256), 0, stream>>>(v1, vth);
  dual_attn_kernel<<<dim3(BB * HH * (LL / LTILE)), dim3(128), 0, stream>>>(q1, q2, k1h, k2h, vth, out);
}
